// BiLSTMEncoder_46462956208396
// MI455X (gfx1250) — compile-verified, matrix operand order checked by reading, not run on hardware
//
#include <hip/hip_runtime.h>
#include <stdint.h>

// ---------------------------------------------------------------------------
// BiLSTM encoder for MI455X (gfx1250, wave32, WMMA).
//   B=64, T=256, D=H=1024, L=2, bidirectional.
// Strategy:
//   * fp16 data / fp32 accumulate via v_wmma_f32_16x16x32_f16
//   * input projections hoisted into big parallel GEMMs (P = X@Wx + b)
//   * persistent recurrence kernel, Wh slice pinned in LDS (128KB/WGP-block)
//     via the Tensor Data Mover (tensor_load_to_lds + s_wait_tensorcnt),
//     register-resident (h,c) state, grid-wide atomic barrier per timestep
//   * double-buffered (software pipelined) fragment loads around every WMMA
// ---------------------------------------------------------------------------

typedef __attribute__((ext_vector_type(16))) _Float16 v16h;
typedef __attribute__((ext_vector_type(8)))  _Float16 v8h;
typedef __attribute__((ext_vector_type(4)))  _Float16 v4h;
typedef __attribute__((ext_vector_type(8)))  float    v8f;
typedef __attribute__((ext_vector_type(4)))  unsigned int u32x4;
typedef __attribute__((ext_vector_type(8)))  int      i32x8;
typedef __attribute__((ext_vector_type(4)))  int      i32x4;

#define B_   64
#define T_   256
#define D_   1024
#define H_   1024
#define G4_  4096      // 4*H (gate width)
#define KT_  32        // K tiles of 32 (K = 1024)
#define NT_  256       // N tiles of 16 over 4H

static __device__ __forceinline__ float sigf(float x) {
  return 1.0f / (1.0f + __expf(-x));
}

static __device__ __forceinline__ v16h cat8(v8h lo, v8h hi) {
  v16h r;
#pragma unroll
  for (int i = 0; i < 8; ++i) { r[i] = lo[i]; r[i + 8] = hi[i]; }
  return r;
}

// Load one A fragment (16x32 f16, row-major source) for this lane.
// ISA 7.12.2: lane = 16*hh + mm holds row M0+mm, K chunks [8*hh..+7], [16+8*hh..+7].
static __device__ __forceinline__ v16h load_a_frag(const _Float16* row_ptr, int kbase) {
  v8h lo = *(const v8h*)(row_ptr + kbase);
  v8h hi = *(const v8h*)(row_ptr + kbase + 16);
  return cat8(lo, hi);
}

// ---------------------------------------------------------------------------
// fp32 -> fp16 elementwise convert (vectorized x4)
// ---------------------------------------------------------------------------
__global__ void k_f32_to_f16(const float* __restrict__ src,
                             _Float16* __restrict__ dst, long n4) {
  long i = (long)blockIdx.x * blockDim.x + threadIdx.x;
  if (i >= n4) return;
  float4 v = ((const float4*)src)[i];
  v4h o; o[0] = (_Float16)v.x; o[1] = (_Float16)v.y;
  o[2] = (_Float16)v.z; o[3] = (_Float16)v.w;
  ((v4h*)dst)[i] = o;
}

// ---------------------------------------------------------------------------
// Pack a (K=1024, N=4096) fp32 weight into per-lane WMMA B-fragment layout:
// tile (nt, kk) -> 32 lanes x 16 f16 contiguous (1KB per tile), nt-major so
// all kk fragments of one n-strip stream contiguously (B striping mirrors the
// documented 16-bit A layout, ISA 7.12.2).
// ---------------------------------------------------------------------------
__global__ void k_pack_w(const float* __restrict__ W, _Float16* __restrict__ Wpk) {
  int tid  = blockIdx.x * blockDim.x + threadIdx.x;   // one thread per (tile,lane)
  int lane = tid & 31;
  int tile = tid >> 5;                                 // nt*KT_ + kk
  if (tile >= NT_ * KT_) return;
  int kk = tile & (KT_ - 1);
  int nt = tile / KT_;
  int lm = lane & 15, lh = lane >> 4;
  _Float16* dst = Wpk + (size_t)tile * 512 + (size_t)lane * 16;
#pragma unroll
  for (int j = 0; j < 16; ++j) {
    int v = j >> 1, pos = j & 1;
    int kl = (v < 4 ? 2 * v : 16 + 2 * (v - 4)) + pos + 8 * lh;   // K within tile
    int K = kk * 32 + kl;
    int N = nt * 16 + lm;
    dst[j] = (_Float16)W[(size_t)K * G4_ + N];
  }
}

// ---------------------------------------------------------------------------
// Input-projection GEMM:  P[row, 0:4096] = A_row @ Wpk + bias
//   rows = T*B ordered (s*64 + b); A row source selected by `mode`:
//     mode 0: fw x   -> x16[(b*T + s)       * D]
//     mode 1: bw x   -> x16[(b*T + (T-1-s)) * D]
//     mode 2: hseq   -> h [(s*64 + b + 64)  * H]   (skip zero prologue step)
// Each wave: 4 M-tiles x 1 N-tile; software-pipelined K loop of 32.
// ---------------------------------------------------------------------------
__global__ void __launch_bounds__(256)
k_gemm_pre(const _Float16* __restrict__ A,
           const _Float16* __restrict__ Bpk,
           const float* __restrict__ bias,
           float* __restrict__ P, int mode) {
  int lane = threadIdx.x & 31;
  int wave = threadIdx.x >> 5;
  long gw  = (long)blockIdx.x * 8 + wave;
  int  ntile  = (int)(gw % NT_);
  int  mblock = (int)(gw / NT_);          // 0..255 (64 rows each)
  int lm = lane & 15, lh = lane >> 4;

  const _Float16* arow[4];
#pragma unroll
  for (int mt = 0; mt < 4; ++mt) {
    int row = mblock * 64 + mt * 16 + lm;
    int s = row >> 6, b = row & 63;
    long off;
    if      (mode == 0) off = ((long)b * T_ + s) * D_;
    else if (mode == 1) off = ((long)b * T_ + (T_ - 1 - s)) * D_;
    else                off = ((long)(s * 64 + b) + 64) * H_;
    arow[mt] = A + off;
  }

  v8f acc[4];
#pragma unroll
  for (int mt = 0; mt < 4; ++mt)
#pragma unroll
    for (int r = 0; r < 8; ++r) acc[mt][r] = 0.0f;

  const _Float16* bbase = Bpk + (size_t)ntile * KT_ * 512 + (size_t)lane * 16;
  auto loadB = [&](int kk) -> v16h {
    const v8h* bp = (const v8h*)(bbase + (size_t)kk * 512);
    return cat8(bp[0], bp[1]);
  };

  // software pipeline: fragments for iteration kk+1 are in flight while the
  // WMMAs for iteration kk execute.
  v16h bf = loadB(0);
  v16h af[4];
#pragma unroll
  for (int mt = 0; mt < 4; ++mt) af[mt] = load_a_frag(arow[mt], lh * 8);

#pragma unroll 2
  for (int kk = 0; kk < KT_; ++kk) {
    int kn = (kk + 1) & (KT_ - 1);
    v16h bfn = loadB(kn);
    v16h afn[4];
    int kbn = kn * 32 + lh * 8;
#pragma unroll
    for (int mt = 0; mt < 4; ++mt) afn[mt] = load_a_frag(arow[mt], kbn);
    __builtin_prefetch(bbase + (size_t)((kk + 2) & (KT_ - 1)) * 512, 0, 3);
#pragma unroll
    for (int mt = 0; mt < 4; ++mt)
      acc[mt] = __builtin_amdgcn_wmma_f32_16x16x32_f16(
          false, af[mt], false, bf, (short)0, acc[mt], false, false);
    bf = bfn;
#pragma unroll
    for (int mt = 0; mt < 4; ++mt) af[mt] = afn[mt];
  }

  float bv = bias[ntile * 16 + lm];
#pragma unroll
  for (int mt = 0; mt < 4; ++mt)
#pragma unroll
    for (int r = 0; r < 8; ++r) {
      int row = mblock * 64 + mt * 16 + r + 8 * lh;   // C layout: M = r (+8 hi half)
      P[(long)row * G4_ + ntile * 16 + lm] = acc[mt][r] + bv;
    }
}

// ---------------------------------------------------------------------------
// Persistent recurrence kernel: one launch runs all T steps of one (dir,layer).
// Grid: 64 blocks (one per 16-column group of H) x 128 threads (4 waves, one
// per 16-row M tile of the 64-batch). Block pins its Wh slice (4 gates x 32
// k-tiles x 1KB = 128KB) in LDS, staged by the Tensor Data Mover when
// available. Per step: gates = P_t + h_{t-1} @ Wh via 32x4 WMMAs
// (pipelined), in-register LSTM cell update, publish h_t as fp16, grid-wide
// barrier (monotonic atomic counter + s_sleep spin; 64 co-resident blocks).
// ---------------------------------------------------------------------------
__global__ void __launch_bounds__(128)
k_lstm_rec(const float* __restrict__ P,
           const _Float16* __restrict__ Whpk,
           _Float16* __restrict__ hseq,
           const int* __restrict__ lengths,
           float* __restrict__ out,
           int dir, int last,
           unsigned* __restrict__ bar) {
  extern __shared__ _Float16 lds[];                 // 65536 f16 = 128KB
  const int lane = threadIdx.x & 31;
  const int wave = threadIdx.x >> 5;
  const int ng   = blockIdx.x;                      // 0..63 column group of H
  const int M0   = wave * 16;
  const int lm = lane & 15, lh = lane >> 4;

  // ---- stage Wh fragments into LDS (gate-major, linear) ----
#if __has_builtin(__builtin_amdgcn_tensor_load_to_lds)
  if (wave == 0) {
    // 2-D D#: 4 rows (gates) of 4096 8-byte units (32KB each), global row
    // stride = 64 n-strips, packed contiguously into LDS at offset 0.
    unsigned long long ga =
        (unsigned long long)(const void*)(Whpk + (size_t)ng * KT_ * 512);
    u32x4 g0;
    g0[0] = 1u;                                          // count=1
    g0[1] = 0u;                                          // lds_addr = 0
    g0[2] = (unsigned)(ga & 0xffffffffu);                // global_addr[31:0]
    g0[3] = (unsigned)((ga >> 32) & 0x01ffffffu) | (2u << 30);  // addr[56:32]|type=2
    i32x8 g1;
    g1[0] = (3 << 16);                 // workgroup_mask=0, data_size=3 (8B)
    g1[1] = (int)(4096u << 16);        // tensor_dim0[15:0]=4096
    g1[2] = (int)(4u << 16);           // tensor_dim0 hi=0 | tensor_dim1 lo=4
    g1[3] = (int)(4096u << 16);        // tensor_dim1 hi=0 | tile_dim0=4096
    g1[4] = 4;                         // tile_dim1=4, tile_dim2=0
    g1[5] = 64 * 4096;                 // tensor_dim0_stride lo = 262144
    g1[6] = 0;                         // stride0 hi=0, tensor_dim1_stride lo=0
    g1[7] = 0;
    i32x4 z4 = {0, 0, 0, 0};
#if defined(__clang_major__) && __clang_major__ >= 23
    i32x8 z8 = {0, 0, 0, 0, 0, 0, 0, 0};
    __builtin_amdgcn_tensor_load_to_lds(g0, g1, z4, z4, z8, 0);
#else
    __builtin_amdgcn_tensor_load_to_lds(g0, g1, z4, z4, 0);
#endif
    __builtin_amdgcn_s_wait_tensorcnt(0);
  }
#else
#pragma unroll
  for (int g = 0; g < 4; ++g) {
    const uint4* src = (const uint4*)(Whpk + (size_t)(g * 64 + ng) * KT_ * 512);
    uint4* dst = (uint4*)(lds + (size_t)g * KT_ * 512);
#pragma unroll
    for (int i = 0; i < 16; ++i)                    // 2048 uint4 / 128 threads
      dst[threadIdx.x + i * 128] = src[threadIdx.x + i * 128];
  }
#endif
  __syncthreads();

  // ---- per-lane owned output elements: 8 (m,n) pairs ----
  int mrow[8], len[8];
  const int ncol = ng * 16 + lm;
#pragma unroll
  for (int r = 0; r < 8; ++r) {
    mrow[r] = M0 + r + 8 * lh;                      // batch index (C-layout M)
    len[r]  = lengths[mrow[r]];
  }
  float c[8], h[8];
#pragma unroll
  for (int r = 0; r < 8; ++r) { c[r] = 0.0f; h[r] = 0.0f; }

  const _Float16* arow = hseq + (size_t)(M0 + lm) * H_;   // advanced per step

  auto loadBg = [&](int g, int kk) -> v16h {
    const v8h* bp = (const v8h*)(lds + ((size_t)(g * KT_ + kk) * 32 + lane) * 16);
    return cat8(bp[0], bp[1]);
  };

#pragma unroll 1
  for (int s = 0; s < T_; ++s) {
    // init accumulators from precomputed x-projection P
    v8f acc[4];
#pragma unroll
    for (int g = 0; g < 4; ++g)
#pragma unroll
      for (int r = 0; r < 8; ++r)
        acc[g][r] = P[((long)(s * 64 + mrow[r])) * G4_ + g * H_ + ncol];

    const _Float16* hprev = arow + (size_t)s * B_ * H_;

    // pipelined h_{t-1} @ Wh
    v16h af = load_a_frag(hprev, lh * 8);
    v16h bf[4];
#pragma unroll
    for (int g = 0; g < 4; ++g) bf[g] = loadBg(g, 0);

#pragma unroll 2
    for (int kk = 0; kk < KT_; ++kk) {
      int kn = (kk + 1) & (KT_ - 1);
      v16h afn = load_a_frag(hprev, kn * 32 + lh * 8);
      v16h bfn[4];
#pragma unroll
      for (int g = 0; g < 4; ++g) bfn[g] = loadBg(g, kn);
#pragma unroll
      for (int g = 0; g < 4; ++g)
        acc[g] = __builtin_amdgcn_wmma_f32_16x16x32_f16(
            false, af, false, bf[g], (short)0, acc[g], false, false);
      af = afn;
#pragma unroll
      for (int g = 0; g < 4; ++g) bf[g] = bfn[g];
    }

    // ---- in-register LSTM cell update ----
    _Float16* hnext = hseq + (size_t)(s + 1) * B_ * H_;
#pragma unroll
    for (int r = 0; r < 8; ++r) {
      float iv = sigf(acc[0][r]);
      float fv = sigf(acc[1][r]);
      float gv = tanhf(acc[2][r]);
      float ov = sigf(acc[3][r]);
      float cn = fv * c[r] + iv * gv;
      float hn = ov * tanhf(cn);
      bool  vm = (s < len[r]);
      float h2 = vm ? hn : h[r];
      float c2 = vm ? cn : c[r];
      h[r] = h2; c[r] = c2;
      hnext[(size_t)mrow[r] * H_ + ncol] = (_Float16)h2;
      if (last) {
        int tout = dir ? (T_ - 1 - s) : s;
        out[(size_t)mrow[r] * (T_ * 2 * H_) + (size_t)tout * (2 * H_) + dir * H_ + ncol] = h2;
        if (s == T_ - 1)
          out[(size_t)B_ * T_ * 2 * H_ + (size_t)mrow[r] * (2 * H_) + dir * H_ + ncol] = h2;
      }
    }

    // ---- grid-wide barrier (monotonic; 64 co-resident blocks) ----
    __threadfence();
    __syncthreads();
    if (threadIdx.x == 0) {
      atomicAdd(bar, 1u);
      unsigned target = (unsigned)(s + 1) * 64u;
      while (__hip_atomic_load(bar, __ATOMIC_ACQUIRE, __HIP_MEMORY_SCOPE_AGENT) < target)
        __builtin_amdgcn_s_sleep(1);
    }
    __syncthreads();
  }
}

// ---------------------------------------------------------------------------
// Host launcher
// ---------------------------------------------------------------------------
extern "C" void kernel_launch(void* const* d_in, const int* in_sizes, int n_in,
                              void* d_out, int out_size, void* d_ws, size_t ws_size,
                              hipStream_t stream) {
  const float* x      = (const float*)d_in[0];
  const int*   lens   = (const int*)d_in[1];
  const float* fwWx   = (const float*)d_in[2];
  const float* fwWh   = (const float*)d_in[3];
  const float* fwb    = (const float*)d_in[4];
  const float* bwWx   = (const float*)d_in[5];
  const float* bwWh   = (const float*)d_in[6];
  const float* bwb    = (const float*)d_in[7];
  float* out = (float*)d_out;
  char*  ws  = (char*)d_ws;

  // workspace layout
  const size_t OFF_CNT = 0;                                    // 4 counters
  const size_t OFF_XH  = 256;                                  // x fp16
  const size_t XH_SZ   = (size_t)B_ * T_ * D_ * 2;             // 32 MB
  const size_t OFF_WPK = OFF_XH + XH_SZ;
  const size_t WPK_SZ  = (size_t)D_ * G4_ * 2;                 // 8 MB each
  const size_t OFF_H0  = OFF_WPK + 8 * WPK_SZ;
  const size_t HSEQ_SZ = (size_t)(T_ + 1) * B_ * H_ * 2;       // 33.7 MB
  const size_t OFF_H1  = OFF_H0 + HSEQ_SZ;
  const size_t OFF_P   = OFF_H1 + HSEQ_SZ;                     // 268 MB (reused)

  unsigned*  cnt  = (unsigned*)(ws + OFF_CNT);
  _Float16*  xh   = (_Float16*)(ws + OFF_XH);
  _Float16*  h0   = (_Float16*)(ws + OFF_H0);
  _Float16*  h1   = (_Float16*)(ws + OFF_H1);
  float*     P    = (float*)(ws + OFF_P);
  auto wpk = [&](int dir, int layer, int isWh) -> _Float16* {
    return (_Float16*)(ws + OFF_WPK + (size_t)(((dir * 2 + layer) * 2) + isWh) * WPK_SZ);
  };

  hipMemsetAsync(ws + OFF_CNT, 0, 256, stream);

  // x -> fp16
  {
    long n4 = (long)B_ * T_ * D_ / 4;
    k_f32_to_f16<<<(unsigned)((n4 + 255) / 256), 256, 0, stream>>>(x, xh, n4);
  }
  // pack all 8 weight matrices into WMMA B-fragment layout
  for (int dir = 0; dir < 2; ++dir)
    for (int layer = 0; layer < 2; ++layer) {
      const float* Wx = (dir ? bwWx : fwWx) + (size_t)layer * D_ * G4_;
      const float* Wh = (dir ? bwWh : fwWh) + (size_t)layer * H_ * G4_;
      k_pack_w<<<1024, 256, 0, stream>>>(Wx, wpk(dir, layer, 0));
      k_pack_w<<<1024, 256, 0, stream>>>(Wh, wpk(dir, layer, 1));
    }

  const unsigned GEMM_BLOCKS = (256u * NT_) / 8u;   // 8192
  for (int dir = 0; dir < 2; ++dir) {
    const float* b0 = (dir ? bwb : fwb);
    const float* b1 = b0 + G4_;

    // layer 0: P = x @ Wx0 + b0 ; recurrence -> h0seq
    k_gemm_pre<<<GEMM_BLOCKS, 256, 0, stream>>>(xh, wpk(dir, 0, 0), b0, P, dir ? 1 : 0);
    hipMemsetAsync(h0, 0, (size_t)B_ * H_ * 2, stream);          // zero h_{-1}
    k_lstm_rec<<<64, 128, 131072, stream>>>(P, wpk(dir, 0, 1), h0, lens, out,
                                            dir, /*last=*/0, cnt + (dir * 2 + 0));

    // layer 1: P = h0seq @ Wx1 + b1 ; recurrence -> h1seq + outputs/final
    k_gemm_pre<<<GEMM_BLOCKS, 256, 0, stream>>>(h0, wpk(dir, 1, 0), b1, P, 2);
    hipMemsetAsync(h1, 0, (size_t)B_ * H_ * 2, stream);
    k_lstm_rec<<<64, 128, 131072, stream>>>(P, wpk(dir, 1, 1), h1, lens, out,
                                            dir, /*last=*/1, cnt + (dir * 2 + 1));
  }
}
